// Transformer_1743756722786
// MI455X (gfx1250) — compile-verified, matrix operand order checked by reading, not run on hardware
//
#include <hip/hip_runtime.h>

// ---------------- CDNA5 / gfx1250 WMMA transformer forward ----------------
// All heavy math goes through v_wmma_f32_16x16x32_f16 (fp32 accumulate).
// GEMM: float4 (b128) global staging, fp32->f16 convert into LDS, register
// double-buffering so global_load_b128 of tile k+1 overlaps WMMA on tile k.

typedef _Float16 half_t;
typedef __attribute__((ext_vector_type(16))) _Float16 v16h;
typedef __attribute__((ext_vector_type(8)))  float    v8f;

namespace cfg {
constexpr int B  = 4;
constexpr int S  = 512;
constexpr int D  = 512;
constexpr int H  = 8;
constexpr int DK = 64;
constexpr int DV = 64;
constexpr int L  = 6;
constexpr int DFF = 2048;
constexpr int VD = 32000;
constexpr int BS = B * S; // 2048 rows
}

// ---------------------------------------------------------------------------
// Embedding + positional encoding:  out[b,s,d] = emb[tok[b,s],d] + PE(s,d)
// PE faithful to reference: angle = s / 10000^(2d/D), entry d==0 forced 0,
// row s==0 all zeros, sin on even d / cos on odd d for s>=1.
// ---------------------------------------------------------------------------
__global__ __launch_bounds__(256) void tfx_embed_kernel(
    const int* __restrict__ tok, const float* __restrict__ emb,
    float* __restrict__ out)
{
    using namespace cfg;
    long long i = (long long)blockIdx.x * 256 + threadIdx.x; // [0, B*S*D)
    int d = (int)(i & (D - 1));
    long long bs = i / D;
    int s = (int)(bs & (S - 1));
    int t = tok[bs];
    float pe = 0.0f;
    if (s > 0) {
        float a = (d == 0) ? 0.0f
                           : (float)s * __powf(10000.0f, -2.0f * (float)d / (float)D);
        pe = (d & 1) ? __cosf(a) : __sinf(a);
    }
    out[i] = emb[(long long)t * D + d] + pe;
}

// ---------------------------------------------------------------------------
// Generic batched WMMA GEMM: C = A @ B (optionally B transposed), fp32 in/out,
// f16 WMMA compute. 256 threads = 8 waves (wave32).
//   BM x BN block tile, BK = 64 (two wmma K-steps), per-wave WM x WN.
// Batch index bz decomposes as (bb, hh) with bb = bz / batchH to support
// per-(batch,head) strides for attention GEMMs.
// epi: 0 = none, 1 = +bias, 2 = +bias then leaky_relu(0.01),
//      3 = +residual, 4 = +bias +residual
// ---------------------------------------------------------------------------
template <int BM, int BN, int BK, int WM, int WN>
__global__ __launch_bounds__(256) void tfx_gemm(
    const float* __restrict__ A, const float* __restrict__ Bm,
    float* __restrict__ C,
    const float* __restrict__ bias, const float* __restrict__ res,
    int K, int lda, int ldb, int ldc, int ldr,
    long long sAb, long long sAh,
    long long sBb, long long sBh,
    long long sCb, long long sCh,
    long long sRb,
    int batchH, int transB, int epi)
{
    __shared__ half_t As[BM][BK + 8];      // [m][k]
    __shared__ half_t Bs[BN][BK + 8];      // [n][k]  (transposed store)

    const int tid  = threadIdx.x;
    const int wave = tid >> 5;
    const int lane = tid & 31;

    constexpr int WROWS = BM / WM;
    constexpr int MT = WM / 16;
    constexpr int NT = WN / 16;
    const int wr = wave % WROWS;
    const int wc = wave / WROWS;

    const int bz = blockIdx.z;
    const int bb = bz / batchH;
    const int hh = bz - bb * batchH;

    const float* Ab = A  + bb * sAb + hh * sAh;
    const float* Bb = Bm + bb * sBb + hh * sBh;
    float*       Cb = C  + bb * sCb + hh * sCh;
    const float* Rb = res ? (res + bb * sRb) : nullptr;

    const int rowBase = blockIdx.y * BM;
    const int colBase = blockIdx.x * BN;

    v8f acc[MT][NT] = {};

    const int l15 = lane & 15;
    const int kk0 = (lane >> 4) * 8;       // K-group select per ISA layout

    // ---- float4 (b128) staging, register double-buffered ----
    constexpr int KF = BK / 4;             // float4s per K-row (16)
    constexpr int AF4 = BM * KF / 256;     // float4 loads per thread for A (8)
    constexpr int BF4 = BN * KF / 256;     // float4 loads per thread for B
    float4 aReg[AF4], bReg[BF4];

    auto loadA = [&](int kk) {
        #pragma unroll
        for (int i = 0; i < AF4; ++i) {
            int idx = i * 256 + tid;
            int r  = idx / KF;
            int c4 = (idx % KF) * 4;
            aReg[i] = *reinterpret_cast<const float4*>(
                Ab + (long long)(rowBase + r) * lda + (kk + c4));
        }
    };
    auto storeA = [&]() {
        #pragma unroll
        for (int i = 0; i < AF4; ++i) {
            int idx = i * 256 + tid;
            int r  = idx / KF;
            int c4 = (idx % KF) * 4;
            As[r][c4 + 0] = (half_t)aReg[i].x;
            As[r][c4 + 1] = (half_t)aReg[i].y;
            As[r][c4 + 2] = (half_t)aReg[i].z;
            As[r][c4 + 3] = (half_t)aReg[i].w;
        }
    };
    auto loadB = [&](int kk) {
        if (transB) {
            #pragma unroll
            for (int i = 0; i < BF4; ++i) {
                int idx = i * 256 + tid;
                int r4 = (idx % KF) * 4;
                int c  = idx / KF;
                bReg[i] = *reinterpret_cast<const float4*>(
                    Bb + (long long)(colBase + c) * ldb + (kk + r4));
            }
        } else {
            #pragma unroll
            for (int i = 0; i < BF4; ++i) {
                int idx = i * 256 + tid;
                int c4 = (idx % (BN / 4)) * 4;
                int r  = idx / (BN / 4);
                bReg[i] = *reinterpret_cast<const float4*>(
                    Bb + (long long)(kk + r) * ldb + (colBase + c4));
            }
        }
    };
    auto storeB = [&]() {
        if (transB) {
            #pragma unroll
            for (int i = 0; i < BF4; ++i) {
                int idx = i * 256 + tid;
                int r4 = (idx % KF) * 4;
                int c  = idx / KF;
                Bs[c][r4 + 0] = (half_t)bReg[i].x;
                Bs[c][r4 + 1] = (half_t)bReg[i].y;
                Bs[c][r4 + 2] = (half_t)bReg[i].z;
                Bs[c][r4 + 3] = (half_t)bReg[i].w;
            }
        } else {
            #pragma unroll
            for (int i = 0; i < BF4; ++i) {
                int idx = i * 256 + tid;
                int c4 = (idx % (BN / 4)) * 4;
                int r  = idx / (BN / 4);
                Bs[c4 + 0][r] = (half_t)bReg[i].x;
                Bs[c4 + 1][r] = (half_t)bReg[i].y;
                Bs[c4 + 2][r] = (half_t)bReg[i].z;
                Bs[c4 + 3][r] = (half_t)bReg[i].w;
            }
        }
    };

    loadA(0);
    loadB(0);

    for (int k0 = 0; k0 < K; k0 += BK) {
        storeA();
        storeB();
        __syncthreads();

        const int kn = k0 + BK;
        if (kn < K) {          // prefetch next tile while WMMAs run on LDS
            loadA(kn);
            loadB(kn);
        }

        // ---- two 16x16x32 K-steps per staged tile ----
        #pragma unroll
        for (int s = 0; s < BK; s += 32) {
            v16h afrag[MT], bfrag[NT];
            #pragma unroll
            for (int mt = 0; mt < MT; ++mt) {
                int r = wr * WM + mt * 16 + l15;
                #pragma unroll
                for (int e = 0; e < 8; ++e) {
                    afrag[mt][e]     = As[r][s + kk0 + e];
                    afrag[mt][e + 8] = As[r][s + 16 + kk0 + e];
                }
            }
            #pragma unroll
            for (int nt = 0; nt < NT; ++nt) {
                int c = wc * WN + nt * 16 + l15;
                #pragma unroll
                for (int e = 0; e < 8; ++e) {
                    bfrag[nt][e]     = Bs[c][s + kk0 + e];
                    bfrag[nt][e + 8] = Bs[c][s + 16 + kk0 + e];
                }
            }
            #pragma unroll
            for (int mt = 0; mt < MT; ++mt)
                #pragma unroll
                for (int nt = 0; nt < NT; ++nt)
                    acc[mt][nt] = __builtin_amdgcn_wmma_f32_16x16x32_f16(
                        false, afrag[mt], false, bfrag[nt],
                        (short)0, acc[mt][nt], false, false);
        }

        __syncthreads();
    }

    // ---- epilogue: C/D layout: reg r -> m = r + 8*(lane>=16), n = lane&15 ----
    const int mOff = (lane >> 4) << 3;
    #pragma unroll
    for (int mt = 0; mt < MT; ++mt) {
        #pragma unroll
        for (int nt = 0; nt < NT; ++nt) {
            int n = colBase + wc * WN + nt * 16 + l15;
            float bval = (epi == 1 || epi == 2 || epi == 4) ? bias[n] : 0.0f;
            #pragma unroll
            for (int r = 0; r < 8; ++r) {
                int m = rowBase + wr * WM + mt * 16 + mOff + r;
                float v = acc[mt][nt][r] + bval;
                if (epi == 2) v = (v > 0.0f) ? v : 0.01f * v;   // leaky_relu
                if (epi == 3 || epi == 4) v += Rb[(long long)m * ldr + n];
                Cb[(long long)m * ldc + n] = v;
            }
        }
    }
}

// ---------------------------------------------------------------------------
// Masked softmax over last dim of scores[B,H,S,S].
// Faithful: s = raw/8; masked -> 1e-9 (NOT -inf); then softmax.
// mask(k) = tok[b,k]==0  ||  (causal && k > q)
// ---------------------------------------------------------------------------
__global__ __launch_bounds__(256) void tfx_softmax_kernel(
    float* __restrict__ sc, const int* __restrict__ tokens, int causal)
{
    using namespace cfg;
    const int row = blockIdx.x;            // [0, B*H*S)
    const int q = row & (S - 1);
    const int b = row / (H * S);
    float* r = sc + (long long)row * S;
    const int* tok = tokens + (long long)b * S;
    const int t = threadIdx.x;

    float v0 = r[t]       * 0.125f;
    float v1 = r[t + 256] * 0.125f;
    if (tok[t] == 0       || (causal && t > q))         v0 = 1e-9f;
    if (tok[t + 256] == 0 || (causal && (t + 256) > q)) v1 = 1e-9f;

    __shared__ float red[256];
    red[t] = fmaxf(v0, v1);
    __syncthreads();
    for (int s2 = 128; s2 > 0; s2 >>= 1) {
        if (t < s2) red[t] = fmaxf(red[t], red[t + s2]);
        __syncthreads();
    }
    const float mx = red[0];
    __syncthreads();

    float e0 = __expf(v0 - mx), e1 = __expf(v1 - mx);
    red[t] = e0 + e1;
    __syncthreads();
    for (int s2 = 128; s2 > 0; s2 >>= 1) {
        if (t < s2) red[t] += red[t + s2];
        __syncthreads();
    }
    const float inv = 1.0f / red[0];
    r[t]       = e0 * inv;
    r[t + 256] = e1 * inv;
}

// ---------------------------------------------------------------------------
// LayerNorm over D=512: one 128-thread block per row, 4 elems/thread.
// ---------------------------------------------------------------------------
__global__ __launch_bounds__(128) void tfx_layernorm_kernel(
    const float* __restrict__ in, float* __restrict__ out,
    const float* __restrict__ g, const float* __restrict__ beta)
{
    using namespace cfg;
    const long long row = blockIdx.x;
    const float* p = in + row * D;
    float* o = out + row * D;
    const int t = threadIdx.x;

    float vals[4];
    float s1 = 0.0f, s2 = 0.0f;
    #pragma unroll
    for (int i = 0; i < 4; ++i) {
        float v = p[t + i * 128];
        vals[i] = v;
        s1 += v;
        s2 += v * v;
    }
    __shared__ float rs[128], rq[128];
    rs[t] = s1; rq[t] = s2;
    __syncthreads();
    for (int k = 64; k > 0; k >>= 1) {
        if (t < k) { rs[t] += rs[t + k]; rq[t] += rq[t + k]; }
        __syncthreads();
    }
    const float mean = rs[0] * (1.0f / D);
    const float var  = rq[0] * (1.0f / D) - mean * mean;
    const float inv  = rsqrtf(var + 1e-6f);
    #pragma unroll
    for (int i = 0; i < 4; ++i) {
        int d = t + i * 128;
        o[d] = (vals[i] - mean) * inv * g[d] + beta[d];
    }
}

// ---------------------------------------------------------------------------
// Host-side orchestration
// ---------------------------------------------------------------------------
namespace {

using namespace cfg;

struct GemmArgs {
    const float *A, *B, *bias, *res;
    float *C;
    int M, N, K, lda, ldb, ldc, ldr;
    long long sAb, sAh, sBb, sBh, sCb, sCh, sRb;
    int batches, batchH, transB, epi;
};

inline GemmArgs ga_default() {
    GemmArgs a{};
    a.batches = 1; a.batchH = 1; a.transB = 0; a.epi = 0;
    a.bias = nullptr; a.res = nullptr;
    a.ldr = 0; a.sAb = a.sAh = a.sBb = a.sBh = a.sCb = a.sCh = a.sRb = 0;
    return a;
}

inline void launch_gemm128(hipStream_t st, const GemmArgs& a) {
    dim3 grid(a.N / 128, a.M / 128, a.batches), blk(256);
    tfx_gemm<128, 128, 64, 32, 64><<<grid, blk, 0, st>>>(
        a.A, a.B, a.C, a.bias, a.res, a.K, a.lda, a.ldb, a.ldc, a.ldr,
        a.sAb, a.sAh, a.sBb, a.sBh, a.sCb, a.sCh, a.sRb,
        a.batchH, a.transB, a.epi);
}

inline void launch_gemm64(hipStream_t st, const GemmArgs& a) {
    dim3 grid(a.N / 64, a.M / 128, a.batches), blk(256);
    tfx_gemm<128, 64, 64, 16, 64><<<grid, blk, 0, st>>>(
        a.A, a.B, a.C, a.bias, a.res, a.K, a.lda, a.ldb, a.ldc, a.ldr,
        a.sAb, a.sAh, a.sBb, a.sBh, a.sCb, a.sCh, a.sRb,
        a.batchH, a.transB, a.epi);
}

inline void run_mhsa(hipStream_t st,
                     const float* xq, const float* xkv, float* dst,
                     const int* mask_tokens, int causal,
                     const float* Wq, const float* Wk, const float* Wv,
                     const float* Wo, const float* g, const float* bt,
                     float* q, float* k, float* v,
                     float* scores, float* attn, float* tmp)
{
    GemmArgs a;

    a = ga_default(); a.M = BS; a.N = D; a.K = D;
    a.lda = D; a.ldb = D; a.ldc = D;
    a.A = xq;  a.B = Wq; a.C = q; launch_gemm128(st, a);
    a.A = xkv; a.B = Wk; a.C = k; launch_gemm128(st, a);
    a.A = xkv; a.B = Wv; a.C = v; launch_gemm128(st, a);

    a = ga_default(); a.M = S; a.N = S; a.K = DK;
    a.A = q; a.lda = D; a.sAb = (long long)S * D; a.sAh = DK;
    a.B = k; a.ldb = D; a.sBb = (long long)S * D; a.sBh = DK; a.transB = 1;
    a.C = scores; a.ldc = S;
    a.sCb = (long long)H * S * S; a.sCh = (long long)S * S;
    a.batches = B * H; a.batchH = H;
    launch_gemm128(st, a);

    tfx_softmax_kernel<<<dim3(B * H * S), dim3(256), 0, st>>>(
        scores, mask_tokens, causal);

    a = ga_default(); a.M = S; a.N = DV; a.K = S;
    a.A = scores; a.lda = S;
    a.sAb = (long long)H * S * S; a.sAh = (long long)S * S;
    a.B = v; a.ldb = D; a.sBb = (long long)S * D; a.sBh = DV;
    a.C = attn; a.ldc = D; a.sCb = (long long)S * D; a.sCh = DV;
    a.batches = B * H; a.batchH = H;
    launch_gemm64(st, a);

    a = ga_default(); a.M = BS; a.N = D; a.K = D;
    a.lda = D; a.ldb = D; a.ldc = D;
    a.A = attn; a.B = Wo; a.C = tmp;
    a.res = xq; a.ldr = D; a.epi = 3;
    launch_gemm128(st, a);

    tfx_layernorm_kernel<<<dim3(BS), dim3(128), 0, st>>>(tmp, dst, g, bt);
}

inline void run_ffn(hipStream_t st, const float* x, float* dst,
                    const float* W1, const float* c1,
                    const float* W2, const float* c2,
                    const float* g, const float* bt,
                    float* h, float* tmp)
{
    GemmArgs a;

    a = ga_default(); a.M = BS; a.N = DFF; a.K = D;
    a.lda = D; a.ldb = DFF; a.ldc = DFF;
    a.A = x; a.B = W1; a.C = h; a.bias = c1; a.epi = 2;
    launch_gemm128(st, a);

    a = ga_default(); a.M = BS; a.N = D; a.K = DFF;
    a.lda = DFF; a.ldb = D; a.ldc = D;
    a.A = h; a.B = W2; a.C = tmp; a.bias = c2;
    a.res = x; a.ldr = D; a.epi = 4;
    launch_gemm128(st, a);

    tfx_layernorm_kernel<<<dim3(BS), dim3(128), 0, st>>>(tmp, dst, g, bt);
}

} // anonymous namespace

extern "C" void kernel_launch(void* const* d_in, const int* in_sizes, int n_in,
                              void* d_out, int out_size, void* d_ws, size_t ws_size,
                              hipStream_t stream)
{
    using namespace cfg;

    const int*   enc_tok = (const int*)d_in[0];
    const int*   dec_tok = (const int*)d_in[1];
    const float* emb_enc = (const float*)d_in[2];
    const float* emb_dec = (const float*)d_in[3];
    const float* proj_W  = (const float*)d_in[4];

    const float* encp[12];
    const float* decp[12];
    for (int i = 0; i < 12; ++i) {
        encp[i] = (const float*)d_in[5 + i];
        decp[i] = (const float*)d_in[17 + i];
    }
    const long long strW  = (long long)D * D;
    const long long strV  = D;
    const long long strW1 = (long long)D * DFF;
    const long long strC1 = DFF;
    const long long strW2 = (long long)DFF * D;

    float* ws = (float*)d_ws;
    size_t off = 0;
    auto carve = [&](size_t n) { float* p = ws + off; off += n; return p; };
    const size_t act = (size_t)BS * D;
    float* x      = carve(act);                 // encoder stream / enc_out
    float* y      = carve(act);                 // decoder stream
    float* q      = carve(act);
    float* k      = carve(act);
    float* v      = carve(act);
    float* attn   = carve(act);
    float* tmp    = carve(act);
    float* ffn_h  = carve((size_t)BS * DFF);
    float* scores = carve((size_t)B * H * S * S);
    (void)ws_size; (void)n_in; (void)in_sizes; (void)out_size;

    const long long tot = (long long)BS * D;

    // ---- encoder ----
    tfx_embed_kernel<<<dim3((unsigned)(tot / 256)), dim3(256), 0, stream>>>(
        enc_tok, emb_enc, x);

    for (int l = 0; l < L; ++l) {
        run_mhsa(stream, x, x, x, enc_tok, 0,
                 encp[0] + l * strW, encp[1] + l * strW,
                 encp[2] + l * strW, encp[3] + l * strW,
                 encp[4] + l * strV, encp[5] + l * strV,
                 q, k, v, scores, attn, tmp);
        run_ffn(stream, x, x,
                encp[6] + l * strW1, encp[7] + l * strC1,
                encp[8] + l * strW2, encp[9] + l * strV,
                encp[10] + l * strV, encp[11] + l * strV,
                ffn_h, tmp);
    }

    // ---- decoder ----
    tfx_embed_kernel<<<dim3((unsigned)(tot / 256)), dim3(256), 0, stream>>>(
        dec_tok, emb_dec, y);

    for (int l = 0; l < L; ++l) {
        // self-attention: causal + decoder padding mask
        run_mhsa(stream, y, y, y, dec_tok, 1,
                 decp[0] + l * strW, decp[1] + l * strW,
                 decp[2] + l * strW, decp[3] + l * strW,
                 decp[4] + l * strV, decp[5] + l * strV,
                 q, k, v, scores, attn, tmp);
        // cross-attention: SAME weights (faithful quirk), kv = enc_out
        run_mhsa(stream, y, x, y, enc_tok, 0,
                 decp[0] + l * strW, decp[1] + l * strW,
                 decp[2] + l * strW, decp[3] + l * strW,
                 decp[4] + l * strV, decp[5] + l * strV,
                 q, k, v, scores, attn, tmp);
        run_ffn(stream, y, y,
                decp[6] + l * strW1, decp[7] + l * strC1,
                decp[8] + l * strW2, decp[9] + l * strV,
                decp[10] + l * strV, decp[11] + l * strV,
                ffn_h, tmp);
    }

    // ---- final vocab projection: [BS,D] @ [D,VD] -> d_out ----
    GemmArgs a = ga_default();
    a.M = BS; a.N = VD; a.K = D;
    a.lda = D; a.ldb = VD; a.ldc = VD;
    a.A = y; a.B = proj_W; a.C = (float*)d_out;
    launch_gemm128(stream, a);
}
